// SpectralGatedDeltaNet_8881992368325
// MI455X (gfx1250) — matrix operand order repaired and checked by reading, not compiled, not run on hardware
//
#include <hip/hip_runtime.h>
#include <hip/hip_bf16.h>
#include <math.h>

// ---------------------------------------------------------------------------
// SpectralGatedDeltaNet for MI455X (gfx1250, wave32, WMMA)
//   B=2, T=2048, D=1024, H=8, DK=DV=64, M=8
// Pipeline:
//   K0  cast x -> bf16
//   K1  pack [Wq|Wk|Wv|Wbeta|Wtg|Wg] -> bf16 concat (D x 2176)
//   K2  cast Wo -> bf16
//   G1  WMMA bf16 GEMM: P[4096,2176] = xb @ wcat         (f32 accum)
//       - A tile: ASYNC global->LDS (double buffered, ASYNCcnt)
//       - B tile: VGPR transpose -> LDS ([n][k]) so frags are ds_load_b128
//   K3  per-chain gated delta-rule scan (128 chains), rotary+l2norm inline
//   K4  mode-softmax combine * gate -> ocomb bf16 [4096,512]
//   G2  WMMA bf16 GEMM: d_out[4096,1024] = ocomb @ Wo_bf16
// ---------------------------------------------------------------------------

typedef __attribute__((ext_vector_type(16))) __bf16 v16bf;
typedef __attribute__((ext_vector_type(8)))  __bf16 v8bf;
typedef __attribute__((ext_vector_type(8)))  float  v8f;

#define BQ   2
#define TQ   2048
#define DQ   1024
#define HQ   8
#define DKQ  64
#define DVQ  64
#define MQ   8
#define BT   (BQ*TQ)          // 4096
#define NP   2176             // 512 q | 512 k | 512 v | 64 beta | 64 tg | 512 g
#define COL_Q    0
#define COL_K    512
#define COL_V    1024
#define COL_BETA 1536
#define COL_TG   1600
#define COL_G    1664

// ----------------------- CDNA5 async global->LDS helpers --------------------
// GLOBAL_LOAD_ASYNC_TO_LDS_B128: per active lane, copy 16B from global
// (VADDR 64b) directly into LDS at byte offset VDST, tracked by ASYNCcnt.
__device__ inline void async_load_lds_b128(unsigned lds_off, const void* gptr) {
  asm volatile("global_load_async_to_lds_b128 %0, %1, off"
               :: "v"(lds_off), "v"((unsigned long long)(size_t)gptr)
               : "memory");
}
__device__ inline void wait_asynccnt0() {
  asm volatile("s_wait_asynccnt 0" ::: "memory");
}

// ------------------------------- K0/K2: f32 -> bf16 -------------------------
__global__ __launch_bounds__(256) void cast_bf16_kernel(const float* __restrict__ src,
                                                        __bf16* __restrict__ dst, int n) {
  int g = blockIdx.x * 256 + threadIdx.x;
  if (g < n) dst[g] = (__bf16)src[g];
}

// ------------------------------- K1: pack weights ---------------------------
__global__ __launch_bounds__(256) void pack_weights_kernel(
    const float* __restrict__ Wq, const float* __restrict__ Wk,
    const float* __restrict__ Wv, const float* __restrict__ Wbeta,
    const float* __restrict__ Wtg, const float* __restrict__ Wg,
    __bf16* __restrict__ wcat) {
  int g = blockIdx.x * 256 + threadIdx.x;      // D * NP = 1024*2176
  if (g >= DQ * NP) return;
  int d = g / NP, c = g % NP;
  float v;
  if      (c < COL_K)    v = Wq[d * 512 + c];
  else if (c < COL_V)    v = Wk[d * 512 + (c - COL_K)];
  else if (c < COL_BETA) v = Wv[d * 512 + (c - COL_V)];
  else if (c < COL_TG)   v = Wbeta[d * 64 + (c - COL_BETA)];
  else if (c < COL_G)    v = Wtg[d * 64 + (c - COL_TG)];
  else                   v = Wg[d * 512 + (c - COL_G)];
  wcat[g] = (__bf16)v;
}

// ------------------------------- G1/G2: WMMA bf16 GEMM ----------------------
// C[M,N] (f32, row-major) = A[M,K] (bf16 rm) @ B[K,N] (bf16 rm)
// Block = 256 thr = 8 waves (4x2); block tile 128x64; wave tile 32x32
// (2x2 WMMA 16x16x32 tiles); K-step 32.
//
// Fragment layouts per CDNA5 ISA 7.12.2 (wave32):
//   A 16x32 bf16 : lane L (hi=L>>4, r=L&15) half i -> K = i + 8*hi + (i>=8?8:0)
//                  => two contiguous 8-half runs at K = 8*hi and K = 16+8*hi
//   B 32x16 bf16 : lane L (hi=L>>4, n=L&15) half i -> K = i + 8*hi + (i>=8?8:0)
//                  => with B staged TRANSPOSED in LDS ([n][k]) this is two
//                     contiguous 8-half runs -> 2x ds_load_b128
//   C/D 16x16 f32: VGPR j -> row j + 8*hi, col = L&15

__device__ inline v16bf frag_cat(v8bf lo, v8bf hi) {
  return __builtin_shufflevector(lo, hi, 0, 1, 2, 3, 4, 5, 6, 7,
                                          8, 9, 10, 11, 12, 13, 14, 15);
}

__device__ inline v16bf load_a_frag(const __bf16* rowp, int hi) {
  v8bf lo = *reinterpret_cast<const v8bf*>(rowp + 8 * hi);
  v8bf hh = *reinterpret_cast<const v8bf*>(rowp + 16 + 8 * hi);
  return frag_cat(lo, hh);
}

__device__ inline v16bf load_b_frag(const __bf16* colp, int hi) {
  v8bf lo = *reinterpret_cast<const v8bf*>(colp + 8 * hi);
  v8bf hh = *reinterpret_cast<const v8bf*>(colp + 16 + 8 * hi);
  return frag_cat(lo, hh);
}

__global__ __launch_bounds__(256) void gemm_bf16_wmma_kernel(
    const __bf16* __restrict__ A, const __bf16* __restrict__ Bm,
    float* __restrict__ C, int Mdim, int Ndim, int Kdim) {
  __shared__ __align__(16) __bf16 As[2][128 * 32];  // double-buffered [row][k]
  __shared__ __align__(16) __bf16 Bst[64 * 32];     // transposed: [n][k]

  const int tid  = threadIdx.x;
  const int lane = tid & 31;
  const int wave = tid >> 5;
  const int hi   = (lane >> 4) & 1;
  const int ln   = lane & 15;
  const int row0 = blockIdx.y * 128;
  const int col0 = blockIdx.x * 64;
  const int wr   = wave >> 1;      // 0..3 : 32-row slab
  const int wc   = wave & 1;       // 0..1 : 32-col slab

  v8f acc00 = {0.f,0.f,0.f,0.f,0.f,0.f,0.f,0.f};
  v8f acc01 = acc00, acc10 = acc00, acc11 = acc00;

  // staging index precompute
  const int ar = tid >> 1;              // 0..127 : A row within tile
  const int ac = (tid & 1) * 16;        // 0/16   : A k-segment
  const int bk = (tid >> 4) * 2;        // 0,2,..30 : B k-pair
  const int bn = (tid & 15) * 4;        // 0..60  : B col segment

  const unsigned lds_a0 = (unsigned)(size_t)(void*)&As[0][ar * 32 + ac];
  const unsigned lds_a1 = (unsigned)(size_t)(void*)&As[1][ar * 32 + ac];

  // prologue: async-fill A buffer 0 for k0 = 0
  {
    const __bf16* Ap = A + (size_t)(row0 + ar) * Kdim + ac;
    async_load_lds_b128(lds_a0,     Ap);
    async_load_lds_b128(lds_a0 + 16, Ap + 8);
  }

  int cur = 0;
  for (int k0 = 0; k0 < Kdim; k0 += 32, cur ^= 1) {
    // ---- stage B tile 64 cols x 32 k, TRANSPOSED ([n][k]), VGPR path ----
    {
      const __bf16* Bp0 = Bm + (size_t)(k0 + bk) * Ndim + col0 + bn;
      uint2 r0 = *reinterpret_cast<const uint2*>(Bp0);
      uint2 r1 = *reinterpret_cast<const uint2*>(Bp0 + Ndim);
      const unsigned short* h0 = reinterpret_cast<const unsigned short*>(&r0);
      const unsigned short* h1 = reinterpret_cast<const unsigned short*>(&r1);
#pragma unroll
      for (int i = 0; i < 4; ++i) {
        unsigned int packed = (unsigned int)h0[i] | ((unsigned int)h1[i] << 16);
        *reinterpret_cast<unsigned int*>(&Bst[(bn + i) * 32 + bk]) = packed;
      }
    }
    if (k0 + 32 < Kdim)
      __builtin_prefetch(Bm + (size_t)(k0 + 32 + bk) * Ndim + col0 + bn, 0, 3);

    wait_asynccnt0();     // this wave's A-tile async fills are in LDS
    __syncthreads();      // everyone's A + B staged

    // issue async fill of the *other* A buffer for the next K-step; it
    // overlaps with the fragment loads + 4 WMMAs below (ASYNCcnt waited
    // one iteration later).
    if (k0 + 32 < Kdim) {
      const __bf16* Ap = A + (size_t)(row0 + ar) * Kdim + (k0 + 32) + ac;
      unsigned dstoff = cur ? lds_a0 : lds_a1;
      async_load_lds_b128(dstoff,      Ap);
      async_load_lds_b128(dstoff + 16, Ap + 8);
    }

    // ---- fragments: all wide LDS loads (ds_load_b128) ----
    const __bf16* Ab = &As[cur][0];
    v16bf a0 = load_a_frag(Ab + (wr * 32 + ln) * 32, hi);
    v16bf a1 = load_a_frag(Ab + (wr * 32 + 16 + ln) * 32, hi);
    v16bf b0 = load_b_frag(&Bst[(wc * 32 + ln) * 32], hi);
    v16bf b1 = load_b_frag(&Bst[(wc * 32 + 16 + ln) * 32], hi);

    acc00 = __builtin_amdgcn_wmma_f32_16x16x32_bf16(false, a0, false, b0, (short)0, acc00, false, false);
    acc01 = __builtin_amdgcn_wmma_f32_16x16x32_bf16(false, a0, false, b1, (short)0, acc01, false, false);
    acc10 = __builtin_amdgcn_wmma_f32_16x16x32_bf16(false, a1, false, b0, (short)0, acc10, false, false);
    acc11 = __builtin_amdgcn_wmma_f32_16x16x32_bf16(false, a1, false, b1, (short)0, acc11, false, false);
    __syncthreads();      // Bst reused next iteration
  }

  // ---- store C: per tile, VGPR j -> row j+8*hi, col = ln ----
  const int rbase = row0 + wr * 32;
  const int cbase = col0 + wc * 32;
#pragma unroll
  for (int j = 0; j < 8; ++j) {
    int r0r = rbase + j + 8 * hi;
    C[(size_t)r0r * Ndim + cbase + ln]             = acc00[j];
    C[(size_t)r0r * Ndim + cbase + 16 + ln]        = acc01[j];
    C[(size_t)(r0r + 16) * Ndim + cbase + ln]      = acc10[j];
    C[(size_t)(r0r + 16) * Ndim + cbase + 16 + ln] = acc11[j];
  }
}

// ------------------------------- K3: gated delta-rule scan ------------------
// One block per chain (m,b,h); state S[64][64] f32 resident in LDS.
__global__ __launch_bounds__(256) void scan_kernel(
    const float* __restrict__ P,              // [BT][NP]
    const float* __restrict__ log_decay,      // [H][M]
    const float* __restrict__ omega_log_scale,// [H][M]
    float* __restrict__ ys) {                 // [M][B][H][T][DV]
  const int bid = blockIdx.x;                 // 0..127
  const int m = bid >> 4;
  const int b = (bid >> 3) & 1;
  const int h = bid & 7;
  const int tid = threadIdx.x;

  __shared__ float S[DKQ][DVQ + 1];
  __shared__ float qrot[DKQ], krot[DKQ], vv[DVQ];
  __shared__ float part[4][DVQ];
  __shared__ float red[2];
  __shared__ float dvv[DVQ];
  __shared__ float scal[2];                   // decay, beta

  for (int i = tid; i < DKQ * DVQ; i += 256) S[i >> 6][i & 63] = 0.f;

  const float ld    = log_decay[h * MQ + m];
  const float rho   = expf(-log1pf(expf(ld)));        // exp(-softplus(ld))
  const float omega = expf(omega_log_scale[h * MQ + m]);
  // per-thread rotary frequency (used by tid in [64,96))
  float invf = 0.f;
  if (tid >= 64 && tid < 96) {
    int i = tid - 64;
    invf = expf(-(float)i * (logf(10000.0f) / 32.0f)); // 10000^(-i/32)
  }
  __syncthreads();

  const size_t ys_base = ((((size_t)m * BQ + b) * HQ + h) * TQ) * DVQ;

  for (int t = 0; t < TQ; ++t) {
    const float* Prow = P + (size_t)(b * TQ + t) * NP;

    // phase 1: scalars, v load, rotary on q/k
    if (tid == 0) {
      float beta = 1.f / (1.f + expf(-Prow[COL_BETA + h * MQ + m]));
      float tg   = 1.f / (1.f + expf(-Prow[COL_TG   + h * MQ + m]));
      float dec  = tg * rho;
      if (dec < 1e-6f) dec = 1e-6f;                    // exp(log(clip(.)))
      scal[0] = dec; scal[1] = beta;
    }
    if (tid < 64) vv[tid] = Prow[COL_V + h * DVQ + tid];
    if (tid >= 64 && tid < 96) {
      int i = tid - 64;
      float ang = (float)t * omega * invf;
      float s, c;
      __sincosf(ang, &s, &c);
      float q1 = Prow[COL_Q + h * DKQ + i], q2 = Prow[COL_Q + h * DKQ + 32 + i];
      qrot[i] = q1 * c + q2 * s;  qrot[i + 32] = -q1 * s + q2 * c;
      float k1 = Prow[COL_K + h * DKQ + i], k2 = Prow[COL_K + h * DKQ + 32 + i];
      krot[i] = k1 * c + k2 * s;  krot[i + 32] = -k1 * s + k2 * c;
    }
    __syncthreads();

    // phase 2: squared partials for l2norm
    if (tid < 32)                 part[0][tid] = qrot[tid] * qrot[tid] + qrot[tid + 32] * qrot[tid + 32];
    else if (tid < 64) { int i = tid - 32; part[1][i] = krot[i] * krot[i] + krot[i + 32] * krot[i + 32]; }
    __syncthreads();

    // phase 3: finalize norms (q also gets DK^-0.5 scale)
    if (tid == 0) { float s = 1e-6f; for (int i = 0; i < 32; ++i) s += part[0][i]; red[0] = rsqrtf(s) * 0.125f; }
    if (tid == 1) { float s = 1e-6f; for (int i = 0; i < 32; ++i) s += part[1][i]; red[1] = rsqrtf(s); }
    __syncthreads();

    // phase 4: normalize q/k; decay state
    if (tid < 64) { qrot[tid] *= red[0]; krot[tid] *= red[1]; }
    {
      float dec = scal[0];
#pragma unroll
      for (int j = 0; j < 16; ++j) {
        int idx = tid + j * 256;
        S[idx >> 6][idx & 63] *= dec;
      }
    }
    __syncthreads();

    // phase 5: kv = k . S  (partial over 16-row k-groups)
    {
      int v = tid & 63, kg = tid >> 6;
      float s = 0.f;
#pragma unroll
      for (int j = 0; j < 16; ++j) s += krot[kg * 16 + j] * S[kg * 16 + j][v];
      part[kg][v] = s;
    }
    __syncthreads();

    // phase 6: dv = (v - kv) * beta
    if (tid < 64) {
      float kv = part[0][tid] + part[1][tid] + part[2][tid] + part[3][tid];
      dvv[tid] = (vv[tid] - kv) * scal[1];
    }
    __syncthreads();

    // phase 7: rank-1 update  S += k (outer) dv
#pragma unroll
    for (int j = 0; j < 16; ++j) {
      int idx = tid + j * 256;
      int k = idx >> 6, v = idx & 63;
      S[k][v] += krot[k] * dvv[v];
    }
    __syncthreads();

    // phase 8: o = q . S (partials)
    {
      int v = tid & 63, kg = tid >> 6;
      float s = 0.f;
#pragma unroll
      for (int j = 0; j < 16; ++j) s += qrot[kg * 16 + j] * S[kg * 16 + j][v];
      part[kg][v] = s;
    }
    __syncthreads();

    // phase 9: write o_t
    if (tid < 64) {
      float o = part[0][tid] + part[1][tid] + part[2][tid] + part[3][tid];
      ys[ys_base + (size_t)t * DVQ + tid] = o;
    }
    __syncthreads();
  }
}

// ------------------------------- K4: mode combine + gate --------------------
__global__ __launch_bounds__(256) void combine_kernel(
    const float* __restrict__ ys,             // [M][B][H][T][DV]
    const float* __restrict__ P,              // [BT][NP] (gate slice)
    const float* __restrict__ mode_logits,    // [H][M]
    __bf16* __restrict__ ocomb) {             // [BT][H*DV]
  int g = blockIdx.x * 256 + threadIdx.x;     // BT*H*DV = 2097152
  if (g >= BT * HQ * DVQ) return;
  int v = g & 63;
  int tmp = g >> 6;
  int h = tmp & 7;
  int tmp2 = tmp >> 3;
  int t = tmp2 & (TQ - 1);
  int b = tmp2 >> 11;

  // softmax over modes for this head
  float w[MQ];
  {
    float mx = -1e30f;
    for (int i = 0; i < MQ; ++i) { float z = mode_logits[h * MQ + i]; if (z > mx) mx = z; }
    float s = 0.f;
    for (int i = 0; i < MQ; ++i) { w[i] = expf(mode_logits[h * MQ + i] - mx); s += w[i]; }
    float inv = 1.f / s;
    for (int i = 0; i < MQ; ++i) w[i] *= inv;
  }
  float o = 0.f;
  for (int mm = 0; mm < MQ; ++mm) {
    size_t idx = ((((size_t)mm * BQ + b) * HQ + h) * TQ + t) * DVQ + v;
    o += w[mm] * ys[idx];
  }
  float graw = P[(size_t)(b * TQ + t) * NP + COL_G + h * DVQ + v];
  float gate = 1.f / (1.f + expf(-graw));
  ocomb[(size_t)(b * TQ + t) * (HQ * DVQ) + h * DVQ + v] = (__bf16)(o * gate);
}

// ------------------------------- launcher -----------------------------------
extern "C" void kernel_launch(void* const* d_in, const int* in_sizes, int n_in,
                              void* d_out, int out_size, void* d_ws, size_t ws_size,
                              hipStream_t stream) {
  (void)in_sizes; (void)n_in; (void)out_size; (void)ws_size;
  const float* x      = (const float*)d_in[0];
  const float* Wq     = (const float*)d_in[1];
  const float* Wk     = (const float*)d_in[2];
  const float* Wv     = (const float*)d_in[3];
  const float* Wbeta  = (const float*)d_in[4];
  const float* Wtg    = (const float*)d_in[5];
  const float* mlog   = (const float*)d_in[6];
  const float* ldec   = (const float*)d_in[7];
  const float* olog   = (const float*)d_in[8];
  const float* Wg     = (const float*)d_in[9];
  const float* Wo     = (const float*)d_in[10];
  float* out = (float*)d_out;

  char* ws = (char*)d_ws;
  // 256B-aligned workspace carve-up (~116 MB total)
  __bf16* xb    = (__bf16*)(ws + 0);                    // 4096*1024  bf16 ->  8388608 B
  __bf16* wcat  = (__bf16*)(ws + 8388608);              // 1024*2176  bf16 ->  4456448 B
  __bf16* wob   = (__bf16*)(ws + 12845056);             //  512*1024  bf16 ->  1048576 B
  float*  P     = (float*)(ws + 13893632);              // 4096*2176  f32  -> 35651584 B
  float*  ys    = (float*)(ws + 49545216);              // 8*2*8*2048*64 f32 -> 67108864 B
  __bf16* ocomb = (__bf16*)(ws + 116654080);            // 4096*512   bf16 ->  4194304 B

  // K0: x -> bf16
  cast_bf16_kernel<<<(BT * DQ + 255) / 256, 256, 0, stream>>>(x, xb, BT * DQ);
  // K1: pack projection weights -> bf16 concat
  pack_weights_kernel<<<(DQ * NP + 255) / 256, 256, 0, stream>>>(Wq, Wk, Wv, Wbeta, Wtg, Wg, wcat);
  // K2: Wo -> bf16
  cast_bf16_kernel<<<(512 * 1024 + 255) / 256, 256, 0, stream>>>(Wo, wob, 512 * 1024);

  // G1: fused projection GEMM  P = xb @ wcat   (4096 x 2176 x 1024)
  gemm_bf16_wmma_kernel<<<dim3(NP / 64, BT / 128), 256, 0, stream>>>(xb, wcat, P, BT, NP, DQ);

  // K3: delta-rule scan over all 128 (m,b,h) chains
  scan_kernel<<<MQ * BQ * HQ, 256, 0, stream>>>(P, ldec, olog, ys);

  // K4: mode combine * gate -> bf16
  combine_kernel<<<(BT * HQ * DVQ + 255) / 256, 256, 0, stream>>>(ys, P, mlog, ocomb);

  // G2: output GEMM  out = ocomb @ Wo   (4096 x 1024 x 512)
  gemm_bf16_wmma_kernel<<<dim3(DQ / 64, BT / 128), 256, 0, stream>>>(ocomb, wob, out, BT, DQ, 512);
}
